// SwinTransformerBlock_19069654794849
// MI455X (gfx1250) — matrix operand order repaired and checked by reading, not compiled, not run on hardware
//
#include <hip/hip_runtime.h>
#include <hip/hip_bf16.h>

typedef _Float16 f16;
typedef __attribute__((ext_vector_type(16))) _Float16 v16h;
typedef __attribute__((ext_vector_type(8)))  float    v8f;

#define DEVINL __device__ __forceinline__

static constexpr int Bb = 16, Hh = 112, Ww = 112, Cc = 128;
static constexpr int WS = 7, SS = 3, NH = 4, HD = 32;
static constexpr int NWIN = 4096;           // 16 * 16 * 16 windows
static constexpr int NPAD = 64;             // padded tokens per window
static constexpr int NTOK = 49;             // 7*7
static constexpr int ROWS_P = NWIN * NPAD;  // 262144 padded rows
static constexpr int ROWS_R = Bb * Hh * Ww; // 200704 raster rows

// ---------- fragment helpers (wave32; A and B use contiguous k = 16*(lane>>4) + s) ----------
DEVINL v8f vzero() { v8f z = {0.f,0.f,0.f,0.f,0.f,0.f,0.f,0.f}; return z; }

// B fragment / generic 16 contiguous halves (32B)
DEVINL v16h ld32(const f16* p) {
    v16h r;
    ((uint4*)&r)[0] = ((const uint4*)p)[0];
    ((uint4*)&r)[1] = ((const uint4*)p)[1];
    return r;
}
// A fragment: row pointer to a 32-half K-span; chunks at hs*16 and hs*16+8
DEVINL v16h ldA(const f16* row32, int hs) {
    v16h r;
    ((uint4*)&r)[0] = *(const uint4*)(row32 + hs * 16);
    ((uint4*)&r)[1] = *(const uint4*)(row32 + hs * 16 + 8);
    return r;
}
DEVINL v8f wmma16(v16h a, v16h b, v8f c) {
    return __builtin_amdgcn_wmma_f32_16x16x32_f16(false, a, false, b, (short)0, c, false, false);
}
DEVINL int region7(int g) { return g < 105 ? 0 : (g < 109 ? 1 : 2); }

// ---------- kernel 1: pack weights f32 -> f16 B-fragment order ----------
// layout: [nt][kt][lane(32)][i(16)], element = W[kt*32 + (lane>>4)*16 + i][nt*16 + (lane&15)]
__global__ void k_pack(const float* qkv_w, const float* proj_w,
                       const float* fc1_w, const float* fc2_w, f16* pw) {
    int gid = blockIdx.x * 256 + threadIdx.x;
    if (gid >= 196608) return;
    const float* W; int K, N, loc;
    if (gid < 49152)       { W = qkv_w;  K = 128; N = 384; loc = gid; }
    else if (gid < 65536)  { W = proj_w; K = 128; N = 128; loc = gid - 49152; }
    else if (gid < 131072) { W = fc1_w;  K = 128; N = 512; loc = gid - 65536; }
    else                   { W = fc2_w;  K = 512; N = 128; loc = gid - 131072; }
    int i = loc & 15, lane = (loc >> 4) & 31, tile = loc >> 9;
    int nkt = K >> 5;
    int kt = tile % nkt, nt = tile / nkt;
    int n = nt * 16 + (lane & 15);
    int k = kt * 32 + (lane >> 4) * 16 + i;
    pw[gid] = (f16)W[k * N + n];
}

// ---------- kernel 2: LN1 + shift + window partition -> wins f16 [4096][64][128] ----------
__global__ void k_ln1(const float* __restrict__ x, const float* __restrict__ g,
                      const float* __restrict__ b, f16* __restrict__ wins) {
    int lane = threadIdx.x & 31, wv = threadIdx.x >> 5;
    int row = blockIdx.x * 8 + wv;           // 0..262143
    int m = row >> 6, n = row & 63;
    f16* dst = wins + (size_t)row * 128 + lane * 4;
    if (n >= NTOK) { uint2 z = {0u, 0u}; *(uint2*)dst = z; return; }
    int bb = m >> 8, wi = m & 255, wy = wi >> 4, wx = wi & 15;
    int ty = n / 7, tx = n - ty * 7;
    int y = wy * 7 + ty + SS; if (y >= Hh) y -= Hh;
    int xx = wx * 7 + tx + SS; if (xx >= Ww) xx -= Ww;
    const float* src = x + (size_t)((bb * Hh + y) * Ww + xx) * 128;
    float4 v = *(const float4*)(src + lane * 4);
    float s = v.x + v.y + v.z + v.w;
    float q = v.x * v.x + v.y * v.y + v.z * v.z + v.w * v.w;
    for (int off = 1; off < 32; off <<= 1) {
        s += __shfl_xor(s, off, 32);
        q += __shfl_xor(q, off, 32);
    }
    float mean = s * (1.f / 128.f);
    float rstd = rsqrtf(q * (1.f / 128.f) - mean * mean + 1e-5f);
    float4 gg = *(const float4*)(g + lane * 4);
    float4 bv = *(const float4*)(b + lane * 4);
    f16 o[4];
    o[0] = (f16)((v.x - mean) * rstd * gg.x + bv.x);
    o[1] = (f16)((v.y - mean) * rstd * gg.y + bv.y);
    o[2] = (f16)((v.z - mean) * rstd * gg.z + bv.z);
    o[3] = (f16)((v.w - mean) * rstd * gg.w + bv.w);
    *(uint2*)dst = *(uint2*)o;
}

// ---------- kernel 3: QKV GEMM (16 rows x 384) -> q,k [m][h][64][32], vT [m][h][32][64] ----------
__global__ void k_qkv(const f16* __restrict__ wins, const f16* __restrict__ pw,
                      const float* __restrict__ qkv_b,
                      f16* __restrict__ qb, f16* __restrict__ kb, f16* __restrict__ vt) {
    int lane = threadIdx.x & 31, wv = threadIdx.x >> 5; // wave 0=q 1=k 2=v
    int l15 = lane & 15, hs = lane >> 4;
    int r0 = blockIdx.x * 16;
    const f16* arow = wins + (size_t)(r0 + l15) * 128;
    v8f acc[8]; for (int i = 0; i < 8; i++) acc[i] = vzero();
    for (int kt = 0; kt < 4; kt++) {
        v16h a = ldA(arow + kt * 32, hs);
        for (int ntl = 0; ntl < 8; ntl++) {
            int nt = wv * 8 + ntl;
            v16h bf = ld32(pw + ((size_t)(nt * 4 + kt) * 32 + lane) * 16);
            acc[ntl] = wmma16(a, bf, acc[ntl]);
        }
    }
    __shared__ __align__(16) f16 st[3][16][128];
    for (int ntl = 0; ntl < 8; ntl++) {
        int cl = ntl * 16 + l15;
        float bias = qkv_b[wv * 128 + cl];
        for (int e = 0; e < 8; e++) st[wv][e + hs * 8][cl] = (f16)(acc[ntl][e] + bias);
    }
    __syncthreads();
    int m = r0 >> 6, n0 = r0 & 63;
    if (wv < 2) {
        f16* base = (wv == 0) ? qb : kb;
        int n = n0 + l15;
        for (int hh = 0; hh < 2; hh++) {
            int h = hs * 2 + hh;
            f16* d = base + ((size_t)(m * 4 + h) * 64 + n) * 32;
            const f16* s = &st[wv][l15][h * 32];
            for (int c = 0; c < 4; c++) ((uint4*)d)[c] = ((const uint4*)s)[c];
        }
    } else {
        for (int k4 = 0; k4 < 4; k4++) {
            int idx = lane + 32 * k4;
            int h = idx >> 5, dd = idx & 31;
            __align__(16) f16 tmp[16];
            for (int r = 0; r < 16; r++) tmp[r] = st[2][r][idx];
            f16* d = vt + ((size_t)(m * 4 + h) * 32 + dd) * 64 + n0;
            ((uint4*)d)[0] = ((uint4*)tmp)[0];
            ((uint4*)d)[1] = ((uint4*)tmp)[1];
        }
    }
}

// ---------- kernel 4: windowed attention, one wave per (window, head) ----------
__global__ void k_attn(const f16* __restrict__ qb, const f16* __restrict__ kb,
                       const f16* __restrict__ vt, const float* __restrict__ rel,
                       f16* __restrict__ aout) {
    int lane = threadIdx.x & 31, h = threadIdx.x >> 5;
    int l15 = lane & 15, hs = lane >> 4;
    int m = blockIdx.x;
    int wi = m & 255, wy = wi >> 4, wx = wi & 15;
    const f16* q = qb + (size_t)(m * 4 + h) * 64 * 32;
    const f16* k = kb + (size_t)(m * 4 + h) * 64 * 32;
    const f16* v = vt + (size_t)(m * 4 + h) * 32 * 64;
    __shared__ __align__(16) f16 P[4][64][64];
    __shared__ __align__(16) f16 O[4][64][32];

    int tyj[4], txj[4], rj[4]; bool jv[4];
    for (int tj = 0; tj < 4; tj++) {
        int j = tj * 16 + l15;
        jv[tj] = (j < NTOK);
        int ty = j / 7, tx = j - ty * 7;
        tyj[tj] = ty; txj[tj] = tx;
        rj[tj] = region7(wy * 7 + ty) * 4 + region7(wx * 7 + tx);
    }
    const float scale = 0.17677669529663687f; // 1/sqrt(32)

    for (int ti = 0; ti < 4; ti++) {
        v16h a = ldA(q + (size_t)(ti * 16 + l15) * 32, hs);
        v8f sc[4];
        for (int tj = 0; tj < 4; tj++) {
            v16h bf = ld32(k + (size_t)(tj * 16 + l15) * 32 + hs * 16);
            sc[tj] = wmma16(a, bf, vzero());
        }
        // scale + relative-position bias + shift mask
        for (int e = 0; e < 8; e++) {
            int i = ti * 16 + hs * 8 + e;
            bool iv = (i < NTOK);
            int tyi = 0, txi = 0, ri = 0;
            if (iv) {
                tyi = i / 7; txi = i - tyi * 7;
                ri = region7(wy * 7 + tyi) * 4 + region7(wx * 7 + txi);
            }
            for (int tj = 0; tj < 4; tj++) {
                float val = sc[tj][e] * scale;
                if (!jv[tj]) val = -1e30f;
                else if (iv) {
                    int idx = (tyi - tyj[tj] + 6) * 13 + (txi - txj[tj] + 6);
                    val += rel[idx * 4 + h];
                    if (ri != rj[tj]) val -= 100.f;
                }
                sc[tj][e] = val;
            }
        }
        // softmax over 64 cols (16 lanes within half x 4 tiles)
        for (int e = 0; e < 8; e++) {
            float mx = fmaxf(fmaxf(sc[0][e], sc[1][e]), fmaxf(sc[2][e], sc[3][e]));
            for (int off = 1; off < 16; off <<= 1) mx = fmaxf(mx, __shfl_xor(mx, off, 32));
            float p0 = __expf(sc[0][e] - mx), p1 = __expf(sc[1][e] - mx);
            float p2 = __expf(sc[2][e] - mx), p3 = __expf(sc[3][e] - mx);
            float sum = p0 + p1 + p2 + p3;
            for (int off = 1; off < 16; off <<= 1) sum += __shfl_xor(sum, off, 32);
            float inv = 1.f / sum;
            sc[0][e] = p0 * inv; sc[1][e] = p1 * inv;
            sc[2][e] = p2 * inv; sc[3][e] = p3 * inv;
        }
        for (int tj = 0; tj < 4; tj++)
            for (int e = 0; e < 8; e++)
                P[h][ti * 16 + hs * 8 + e][tj * 16 + l15] = (f16)sc[tj][e];
    }
    __syncthreads();
    // out = P(64x64) @ v(64x32), B fragments from vT rows
    for (int ti = 0; ti < 4; ti++) {
        v8f oa[2]; oa[0] = vzero(); oa[1] = vzero();
        for (int kt = 0; kt < 2; kt++) {
            v16h a = ldA(&P[h][ti * 16 + l15][kt * 32], hs);
            for (int nt = 0; nt < 2; nt++) {
                v16h bf = ld32(v + (size_t)(nt * 16 + l15) * 64 + kt * 32 + hs * 16);
                oa[nt] = wmma16(a, bf, oa[nt]);
            }
        }
        for (int nt = 0; nt < 2; nt++)
            for (int e = 0; e < 8; e++)
                O[h][ti * 16 + hs * 8 + e][nt * 16 + l15] = (f16)oa[nt][e];
    }
    __syncthreads();
    for (int rr = lane; rr < 64; rr += 32) {
        const f16* s = &O[h][rr][0];
        f16* d = aout + ((size_t)m * 64 + rr) * 128 + h * 32;
        for (int c = 0; c < 4; c++) ((uint4*)d)[c] = ((const uint4*)s)[c];
    }
}

// ---------- kernel 5: proj GEMM + window-reverse + unshift + residual + LN2 ----------
__global__ void k_proj(const f16* __restrict__ aout, const f16* __restrict__ pw,
                       const float* __restrict__ proj_b, const float* __restrict__ x,
                       const float* __restrict__ g2, const float* __restrict__ b2,
                       float* __restrict__ x1, f16* __restrict__ xn2) {
    int lane = threadIdx.x & 31, l15 = lane & 15, hs = lane >> 4;
    int r0 = blockIdx.x * 16;
    const f16* arow = aout + (size_t)(r0 + l15) * 128;
    const f16* pb = pw + 49152;
    v8f acc[8]; for (int i = 0; i < 8; i++) acc[i] = vzero();
    for (int kt = 0; kt < 4; kt++) {
        v16h a = ldA(arow + kt * 32, hs);
        for (int nt = 0; nt < 8; nt++) {
            v16h bf = ld32(pb + ((size_t)(nt * 4 + kt) * 32 + lane) * 16);
            acc[nt] = wmma16(a, bf, acc[nt]);
        }
    }
    __shared__ __align__(16) float st[16][128];
    for (int nt = 0; nt < 8; nt++) {
        float bias = proj_b[nt * 16 + l15];
        for (int e = 0; e < 8; e++) st[e + hs * 8][nt * 16 + l15] = acc[nt][e] + bias;
    }
    __syncthreads();
    if (lane < 16) {
        int n = (r0 & 63) + lane;
        if (n < NTOK) {
            int m = r0 >> 6;
            int bb = m >> 8, wi = m & 255, wy = wi >> 4, wx = wi & 15;
            int ty = n / 7, tx = n - ty * 7;
            int y = wy * 7 + ty + SS; if (y >= Hh) y -= Hh;
            int xx = wx * 7 + tx + SS; if (xx >= Ww) xx -= Ww;
            size_t gr = (size_t)((bb * Hh + y) * Ww + xx);
            const float* xr = x + gr * 128;
            float* x1r = x1 + gr * 128;
            float s = 0.f, q = 0.f;
            for (int c = 0; c < 128; c++) {
                float vv = st[lane][c] + xr[c];
                x1r[c] = vv; s += vv; q += vv * vv;
            }
            float mean = s * (1.f / 128.f);
            float rstd = rsqrtf(q * (1.f / 128.f) - mean * mean + 1e-5f);
            f16* xo = xn2 + gr * 128;
            for (int c = 0; c < 128; c++)
                xo[c] = (f16)((x1r[c] - mean) * rstd * g2[c] + b2[c]);
        }
    }
}

// ---------- kernel 6: fc1 GEMM + bias + exact GELU -> h f16 [200704][512] ----------
__global__ void k_fc1(const f16* __restrict__ xn2, const f16* __restrict__ pw,
                      const float* __restrict__ fb, f16* __restrict__ hbuf) {
    int lane = threadIdx.x & 31, wv = threadIdx.x >> 5;
    int l15 = lane & 15, hs = lane >> 4;
    int r0 = blockIdx.x * 16;
    const f16* arow = xn2 + (size_t)(r0 + l15) * 128;
    const f16* pb = pw + 65536;
    v8f acc[8]; for (int i = 0; i < 8; i++) acc[i] = vzero();
    for (int kt = 0; kt < 4; kt++) {
        v16h a = ldA(arow + kt * 32, hs);
        for (int ntl = 0; ntl < 8; ntl++) {
            int nt = wv * 8 + ntl;
            v16h bf = ld32(pb + ((size_t)(nt * 4 + kt) * 32 + lane) * 16);
            acc[ntl] = wmma16(a, bf, acc[ntl]);
        }
    }
    __shared__ __align__(16) f16 st[4][16][128];
    for (int ntl = 0; ntl < 8; ntl++) {
        int colg = wv * 128 + ntl * 16 + l15;
        float bias = fb[colg];
        for (int e = 0; e < 8; e++) {
            float vv = acc[ntl][e] + bias;
            float gel = 0.5f * vv * (1.f + erff(vv * 0.70710678118654752f));
            st[wv][e + hs * 8][ntl * 16 + l15] = (f16)gel;
        }
    }
    __syncthreads();
    const f16* s = &st[wv][l15][hs * 64];
    f16* d = hbuf + (size_t)(r0 + l15) * 512 + wv * 128 + hs * 64;
    for (int c = 0; c < 8; c++) ((uint4*)d)[c] = ((const uint4*)s)[c];
}

// ---------- kernel 7: fc2 GEMM (K=512) + bias + residual -> out f32 ----------
__global__ void k_fc2(const f16* __restrict__ hbuf, const f16* __restrict__ pw,
                      const float* __restrict__ fb, const float* __restrict__ x1,
                      float* __restrict__ out) {
    int lane = threadIdx.x & 31, l15 = lane & 15, hs = lane >> 4;
    int r0 = blockIdx.x * 16;
    const f16* arow = hbuf + (size_t)(r0 + l15) * 512;
    const f16* pb = pw + 131072;
    v8f acc[8]; for (int i = 0; i < 8; i++) acc[i] = vzero();
    for (int kt = 0; kt < 16; kt++) {
        v16h a = ldA(arow + kt * 32, hs);
        for (int nt = 0; nt < 8; nt++) {
            v16h bf = ld32(pb + ((size_t)(nt * 16 + kt) * 32 + lane) * 16);
            acc[nt] = wmma16(a, bf, acc[nt]);
        }
    }
    __shared__ __align__(16) float st[16][128];
    for (int nt = 0; nt < 8; nt++) {
        float bias = fb[nt * 16 + l15];
        for (int e = 0; e < 8; e++) st[e + hs * 8][nt * 16 + l15] = acc[nt][e] + bias;
    }
    __syncthreads();
    const float* s = &st[l15][hs * 64];
    size_t base = (size_t)(r0 + l15) * 128 + hs * 64;
    for (int c = 0; c < 16; c++) {
        float4 vv = ((const float4*)s)[c];
        float4 xr = ((const float4*)(x1 + base))[c];
        vv.x += xr.x; vv.y += xr.y; vv.z += xr.z; vv.w += xr.w;
        ((float4*)(out + base))[c] = vv;
    }
}

// ---------- host launcher ----------
extern "C" void kernel_launch(void* const* d_in, const int* in_sizes, int n_in,
                              void* d_out, int out_size, void* d_ws, size_t ws_size,
                              hipStream_t stream) {
    const float* x      = (const float*)d_in[0];
    const float* n1g    = (const float*)d_in[1];
    const float* n1b    = (const float*)d_in[2];
    const float* qkv_w  = (const float*)d_in[3];
    const float* qkv_b  = (const float*)d_in[4];
    const float* rel    = (const float*)d_in[5];
    const float* proj_w = (const float*)d_in[6];
    const float* proj_b = (const float*)d_in[7];
    const float* n2g    = (const float*)d_in[8];
    const float* n2b    = (const float*)d_in[9];
    const float* fc1_w  = (const float*)d_in[10];
    const float* fc1_b  = (const float*)d_in[11];
    const float* fc2_w  = (const float*)d_in[12];
    const float* fc2_b  = (const float*)d_in[13];
    float* out = (float*)d_out;

    char* ws = (char*)d_ws;
    // region0: wins (then reused as attn output), f16 [4096][64][128]
    f16* wins = (f16*)(ws + 0);                        //  67,108,864 B
    f16* aout = wins;                                  //  reuse after qkv
    // region1: q,k,vT (later reused by h)
    f16* qb   = (f16*)(ws + 67108864ull);              //  67,108,864 B
    f16* kb   = (f16*)(ws + 134217728ull);             //  67,108,864 B
    f16* vt   = (f16*)(ws + 201326592ull);             //  67,108,864 B
    f16* hbuf = (f16*)(ws + 67108864ull);              // 205,520,896 B (overlays q/k/vT)
    // region2: x1 f32
    float* x1 = (float*)(ws + 272629760ull);           // 102,760,448 B
    // region3: xn2 f16
    f16* xn2  = (f16*)(ws + 375390208ull);             //  51,380,224 B
    // region4: packed weights f16 (196608 halves)
    f16* pw   = (f16*)(ws + 426770432ull);             //     393,216 B
    // total workspace needed: ~427,163,648 B

    k_pack<<<768, 256, 0, stream>>>(qkv_w, proj_w, fc1_w, fc2_w, pw);
    k_ln1 <<<ROWS_P / 8, 256, 0, stream>>>(x, n1g, n1b, wins);
    k_qkv <<<ROWS_P / 16, 96, 0, stream>>>(wins, pw, qkv_b, qb, kb, vt);
    k_attn<<<NWIN, 128, 0, stream>>>(qb, kb, vt, rel, aout);
    k_proj<<<ROWS_P / 16, 32, 0, stream>>>(aout, pw, proj_b, x, n2g, n2b, x1, xn2);
    k_fc1 <<<ROWS_R / 16, 128, 0, stream>>>(xn2, pw, fc1_b, hbuf);
    k_fc2 <<<ROWS_R / 16, 32, 0, stream>>>(hbuf, pw, fc2_b, x1, out);
}
